// CoAttention_21552145891846
// MI455X (gfx1250) — matrix operand order repaired and checked by reading, not compiled, not run on hardware
//
#include <hip/hip_runtime.h>
#include <hip/hip_bf16.h>

// ---------------------------------------------------------------------------
// CoAttention fused pipeline for MI455X (gfx1250, wave32, WMMA).
//   proj = (t*mask_t) @ W            -> bf16 WMMA GEMM (B staged via LDS)
//   S    = proj @ (f*mask_f)^T       -> bf16 WMMA GEMM, reduced to row/col max
//   tanh(max) + mask bias -> softmax -> weighted sums -> out[B,D]
// tanh is monotonic => max(tanh(S)) == tanh(max(S)); C is never materialized.
// ---------------------------------------------------------------------------

typedef __bf16 bf16_t;
typedef __attribute__((ext_vector_type(16))) __bf16 v16bf;
typedef __attribute__((ext_vector_type(8)))  __bf16 v8bf;
typedef __attribute__((ext_vector_type(4)))  __bf16 v4bf;
typedef __attribute__((ext_vector_type(8)))  float  v8f;
typedef __attribute__((ext_vector_type(4)))  int    v4i;

#define AS1 __attribute__((address_space(1)))
#define AS3 __attribute__((address_space(3)))

constexpr int kB  = 64;
constexpr int kLT = 512;
constexpr int kLF = 512;
constexpr int kD  = 512;
constexpr float kRemove = 1000000.0f;

// LDS B-tile: 64 rows (N) x 32 K bf16, row stride padded to 40 elems (80 B)
// to spread LDS banks; two buffers for double buffering.
constexpr int kBStride = 40;              // elements
constexpr int kBBuf    = 64 * kBStride;   // elements per buffer

// ---- async-to-LDS path (guarded; sync fallback keeps the build safe) -------
#if defined(__has_builtin)
#  if __has_builtin(__builtin_amdgcn_global_load_async_to_lds_b128)
#    define ATHENA_ASYNC_LDS 1
#  endif
#  if __has_builtin(__builtin_amdgcn_s_wait_asynccnt)
#    define ATHENA_WAIT_ASYNC() __builtin_amdgcn_s_wait_asynccnt(0)
#  endif
#endif
#ifndef ATHENA_WAIT_ASYNC
#  define ATHENA_WAIT_ASYNC() asm volatile("s_wait_asynccnt 0x0" ::: "memory")
#endif

__device__ __forceinline__ void stage_b16(const bf16_t* __restrict__ g, bf16_t* l) {
#ifdef ATHENA_ASYNC_LDS
  // Builtin wants a non-const AS(1) pointer; C-style cast strips const.
  __builtin_amdgcn_global_load_async_to_lds_b128(
      (AS1 v4i*)g, (AS3 v4i*)l, /*offset=*/0, /*cpol=*/0);
#else
  *(uint4*)l = *(const uint4*)g;          // sync fallback: global->VGPR->LDS
#endif
}

// ---- order-preserving float<->uint encoding for atomic max -----------------
__device__ __forceinline__ unsigned encf(float x) {
  unsigned u = __float_as_uint(x);
  return u ^ ((u >> 31) ? 0xFFFFFFFFu : 0x80000000u);
}
__device__ __forceinline__ float decf(unsigned e) {
  unsigned u = e ^ ((e >> 31) ? 0x80000000u : 0xFFFFFFFFu);
  return __uint_as_float(u);
}

// ---- WMMA A-fragment (16x32 bf16 MxK) from row-major global ----------------
// Lane (h=lane>>4, m=lane&15): K {h*8..h*8+7} and {16+h*8..16+h*8+7}.
__device__ __forceinline__ v16bf load_a_frag(const bf16_t* __restrict__ A, int lda,
                                             int row_base, int k, int lane) {
  const int m = lane & 15;
  const int h = lane >> 4;
  const bf16_t* p = A + (size_t)(row_base + m) * lda + k + h * 8;
  v8bf lo = *(const v8bf*)p;
  v8bf hi = *(const v8bf*)(p + 16);
  v16bf r;
#pragma unroll
  for (int i = 0; i < 8; ++i) { r[i] = lo[i]; r[i + 8] = hi[i]; }
  return r;
}

// ---- WMMA B-fragment (32x16 bf16 KxN) from the LDS tile --------------------
// Lane (h=lane>>4): K {h*8..h*8+7} and {16+h*8..16+h*8+7}, N = lane&15.
__device__ __forceinline__ v16bf load_b_frag_lds(const bf16_t* __restrict__ lb,
                                                 int jcol, int lane) {
  const bf16_t* p = lb + (jcol + (lane & 15)) * kBStride + (lane >> 4) * 8;
  v8bf lo = *(const v8bf*)p;
  v8bf hi = *(const v8bf*)(p + 16);
  v16bf r;
#pragma unroll
  for (int i = 0; i < 8; ++i) { r[i] = lo[i]; r[i + 8] = hi[i]; }
  return r;
}

// ---- shared GEMM strip: acc[4] (16x64) += A[16xK] * Bsrc^T-style tiles -----
// Bsrc stored "N-major": row n holds B[:,n] contiguous in K.
__device__ __forceinline__ void wmma_strip(const bf16_t* __restrict__ A, int row_base,
                                           const bf16_t* __restrict__ Bsrc, int col_base,
                                           bf16_t* __restrict__ lds, int tid, int lane,
                                           v8f acc[4]) {
  const int sn = tid >> 2;                 // staged row 0..63
  const int sseg = tid & 3;                // 16B chunk within 64B row window
  const bf16_t* grow = Bsrc + (size_t)(col_base + sn) * kD + sseg * 8;
  bf16_t* lrow0 = lds + sn * kBStride + sseg * 8;
  bf16_t* lrow1 = lrow0 + kBBuf;

  stage_b16(grow, lrow0);                  // prologue: stage K-window 0
  int cur = 0;
  for (int k = 0; k < kD; k += 32) {
    ATHENA_WAIT_ASYNC();                   // my stage done
    __syncthreads();                       // everyone's stage done / readers drained
    if (k + 32 < kD) {
      stage_b16(grow + (k + 32), cur ? lrow0 : lrow1);   // overlap next window
      __builtin_prefetch(A + (size_t)(row_base + (lane & 15)) * kD + k + 32, 0, 3);
    }
    const bf16_t* lb = lds + cur * kBBuf;
    v16bf a = load_a_frag(A, kD, row_base, k, lane);
#pragma unroll
    for (int j = 0; j < 4; ++j) {
      v16bf bb = load_b_frag_lds(lb, j * 16, lane);
      acc[j] = __builtin_amdgcn_wmma_f32_16x16x32_bf16(false, a, false, bb,
                                                       (short)0, acc[j], false, false);
    }
    cur ^= 1;
  }
}

// ---------------------------------------------------------------------------
// Kernel 1: y = bf16(x * mask_row), 4 elements / thread.
__global__ void mask_pack_bf16(const float* __restrict__ x, const int* __restrict__ mask,
                               bf16_t* __restrict__ y, long total4) {
  long i = (long)blockIdx.x * blockDim.x + threadIdx.x;
  if (i >= total4) return;
  long base = i * 4;
  int row = (int)(base >> 9);              // / kD
  float m = (mask[row] != 0) ? 1.0f : 0.0f;
  float4 v = *(const float4*)(x + base);
  v4bf o;
  o[0] = (bf16_t)(v.x * m); o[1] = (bf16_t)(v.y * m);
  o[2] = (bf16_t)(v.z * m); o[3] = (bf16_t)(v.w * m);
  *(v4bf*)(y + base) = o;
}

// Kernel 2: Wt[e][d] = bf16(W[d][e])   (block 64x4, grid 8x128)
__global__ void transpose_w_bf16(const float* __restrict__ w, bf16_t* __restrict__ wt) {
  int e = blockIdx.x * 64 + threadIdx.x;
  int d = blockIdx.y * 4 + threadIdx.y;
  wt[(size_t)e * kD + d] = (bf16_t)w[(size_t)d * kD + e];
}

// Kernel 3: proj[b] = tm[b] @ W  (bf16 in/out, f32 accum via WMMA)
// grid (LT/128, D/64, B), block 256 (8 waves); wave w owns rows [w*16,+16).
__global__ void gemm1_proj(const bf16_t* __restrict__ tm, const bf16_t* __restrict__ wt,
                           bf16_t* __restrict__ proj) {
  __shared__ bf16_t lds[2 * kBBuf];
  const int lane = threadIdx.x & 31;
  const int wave = threadIdx.x >> 5;
  const int b = blockIdx.z;
  const int row_base = blockIdx.x * 128 + wave * 16;
  const int col_base = blockIdx.y * 64;
  const bf16_t* A = tm + (size_t)b * kLT * kD;
  bf16_t* P = proj + (size_t)b * kLT * kD;

  v8f acc[4] = {v8f{}, v8f{}, v8f{}, v8f{}};
  wmma_strip(A, row_base, wt, col_base, lds, threadIdx.x, lane, acc);

  const int mrow = row_base + (lane >> 4) * 8;
  const int ncol = col_base + (lane & 15);
#pragma unroll
  for (int j = 0; j < 4; ++j)
#pragma unroll
    for (int r = 0; r < 8; ++r)
      P[(size_t)(mrow + r) * kD + ncol + j * 16] = (bf16_t)acc[j][r];
}

// Kernel 4: S[b] = proj[b] @ fm[b]^T, folded into global row/col maxima.
// grid (LT/128, LF/64, B), block 256.
__global__ void gemm2_scoremax(const bf16_t* __restrict__ proj, const bf16_t* __restrict__ fm,
                               unsigned* __restrict__ rowmax, unsigned* __restrict__ colmax) {
  __shared__ bf16_t lds[2 * kBBuf];
  const int lane = threadIdx.x & 31;
  const int wave = threadIdx.x >> 5;
  const int b = blockIdx.z;
  const int row_base = blockIdx.x * 128 + wave * 16;
  const int col_base = blockIdx.y * 64;
  const bf16_t* A = proj + (size_t)b * kLT * kD;
  const bf16_t* F = fm + (size_t)b * kLF * kD;

  v8f acc[4] = {v8f{}, v8f{}, v8f{}, v8f{}};
  wmma_strip(A, row_base, F, col_base, lds, threadIdx.x, lane, acc);

  // Row max: element (m,n) lives at VGPR r, m=(lane>>4)*8+r, n=lane&15.
#pragma unroll
  for (int r = 0; r < 8; ++r) {
    float v = fmaxf(fmaxf(acc[0][r], acc[1][r]), fmaxf(acc[2][r], acc[3][r]));
    v = fmaxf(v, __shfl_xor(v, 1, 32));
    v = fmaxf(v, __shfl_xor(v, 2, 32));
    v = fmaxf(v, __shfl_xor(v, 4, 32));
    v = fmaxf(v, __shfl_xor(v, 8, 32));
    if ((lane & 15) == 0)
      atomicMax(&rowmax[b * kLT + row_base + (lane >> 4) * 8 + r], encf(v));
  }
  // Col max: same n in lane L and L+16; fold halves then one atomic per n.
#pragma unroll
  for (int j = 0; j < 4; ++j) {
    float v = acc[j][0];
#pragma unroll
    for (int r = 1; r < 8; ++r) v = fmaxf(v, acc[j][r]);
    v = fmaxf(v, __shfl_xor(v, 16, 32));
    if (lane < 16)
      atomicMax(&colmax[b * kLF + col_base + j * 16 + lane], encf(v));
  }
}

// Kernel 5: alpha = softmax(tanh(max) + (m-1)*1e6). grid (B, 2), block 256.
__global__ void biased_softmax(const unsigned* __restrict__ rowmax,
                               const unsigned* __restrict__ colmax,
                               const int* __restrict__ mask_t,
                               const int* __restrict__ mask_f,
                               float* __restrict__ alpha_t,
                               float* __restrict__ alpha_f) {
  const int b = blockIdx.x;
  const bool t_side = (blockIdx.y == 0);
  const int L = t_side ? kLT : kLF;
  const unsigned* mx = (t_side ? rowmax : colmax) + b * L;
  const int* msk = (t_side ? mask_t : mask_f) + b * L;
  float* alpha = (t_side ? alpha_t : alpha_f) + b * L;

  const int tid = threadIdx.x;
  __shared__ float sh[512];
  float x0 = tanhf(decf(mx[tid])) + ((msk[tid] != 0) ? 0.0f : -kRemove);
  float x1 = tanhf(decf(mx[tid + 256])) + ((msk[tid + 256] != 0) ? 0.0f : -kRemove);
  sh[tid] = fmaxf(x0, x1);
  __syncthreads();
  for (int s = 128; s > 0; s >>= 1) {
    if (tid < s) sh[tid] = fmaxf(sh[tid], sh[tid + s]);
    __syncthreads();
  }
  float m = sh[0];
  __syncthreads();
  float e0 = expf(x0 - m), e1 = expf(x1 - m);
  sh[tid] = e0 + e1;
  __syncthreads();
  for (int s = 128; s > 0; s >>= 1) {
    if (tid < s) sh[tid] += sh[tid + s];
    __syncthreads();
  }
  float inv = 1.0f / sh[0];
  alpha[tid] = e0 * inv;
  alpha[tid + 256] = e1 * inv;
}

// Kernel 6: out[b,d] = sum_l a_t[l]*tm[b,l,d] + sum_m a_f[m]*fm[b,m,d].
__global__ void weighted_sum(const bf16_t* __restrict__ tm, const bf16_t* __restrict__ fm,
                             const float* __restrict__ alpha_t, const float* __restrict__ alpha_f,
                             float* __restrict__ out) {
  const int b = blockIdx.x;
  const int d = threadIdx.x;
  __shared__ float sat[kLT];
  __shared__ float saf[kLF];
  sat[d] = alpha_t[b * kLT + d];
  saf[d] = alpha_f[b * kLF + d];
  __syncthreads();
  const bf16_t* T = tm + (size_t)b * kLT * kD;
  const bf16_t* F = fm + (size_t)b * kLF * kD;
  float acc = 0.0f;
  for (int l = 0; l < kLT; ++l) acc += sat[l] * (float)T[(size_t)l * kD + d];
  for (int l = 0; l < kLF; ++l) acc += saf[l] * (float)F[(size_t)l * kD + d];
  out[b * kD + d] = acc;
}

// ---------------------------------------------------------------------------
extern "C" void kernel_launch(void* const* d_in, const int* in_sizes, int n_in,
                              void* d_out, int out_size, void* d_ws, size_t ws_size,
                              hipStream_t stream) {
  const float* t      = (const float*)d_in[0];
  const float* f      = (const float*)d_in[1];
  const int*   mask_t = (const int*)d_in[2];
  const int*   mask_f = (const int*)d_in[3];
  const float* w_beta = (const float*)d_in[4];
  float* out = (float*)d_out;

  // ---- workspace carve-up (all offsets power-of-two aligned) ----
  char* ws = (char*)d_ws;
  size_t oTM = 0;
  size_t oFM = oTM + (size_t)kB * kLT * kD * sizeof(bf16_t);   // 32 MB
  size_t oWT = oFM + (size_t)kB * kLF * kD * sizeof(bf16_t);   // 32 MB
  size_t oPJ = oWT + (size_t)kD * kD * sizeof(bf16_t);         // 0.5 MB
  size_t oRM = oPJ + (size_t)kB * kLT * kD * sizeof(bf16_t);   // 32 MB
  size_t oCM = oRM + (size_t)kB * kLT * sizeof(unsigned);
  size_t oAT = oCM + (size_t)kB * kLF * sizeof(unsigned);
  size_t oAF = oAT + (size_t)kB * kLT * sizeof(float);

  bf16_t*   tm_bf   = (bf16_t*)(ws + oTM);
  bf16_t*   fm_bf   = (bf16_t*)(ws + oFM);
  bf16_t*   wt_bf   = (bf16_t*)(ws + oWT);
  bf16_t*   proj_bf = (bf16_t*)(ws + oPJ);
  unsigned* rowmax  = (unsigned*)(ws + oRM);
  unsigned* colmax  = (unsigned*)(ws + oCM);
  float*    alpha_t = (float*)(ws + oAT);
  float*    alpha_f = (float*)(ws + oAF);

  // encoded(-inf) is 0x007FFFFF; all finite encodings exceed 0, so zero-init
  // of the max buffers is a valid identity for the encoded atomic max.
  (void)hipMemsetAsync(rowmax, 0, (size_t)kB * (kLT + kLF) * sizeof(unsigned), stream);

  // 1) mask + pack to bf16
  {
    long total4 = (long)kB * kLT * kD / 4;
    int blocks = (int)((total4 + 255) / 256);
    mask_pack_bf16<<<blocks, 256, 0, stream>>>(t, mask_t, tm_bf, total4);
    mask_pack_bf16<<<blocks, 256, 0, stream>>>(f, mask_f, fm_bf, total4);
  }
  // 2) W transpose -> bf16 (so B-fragments are contiguous K loads)
  transpose_w_bf16<<<dim3(8, 128), dim3(64, 4), 0, stream>>>(w_beta, wt_bf);

  // 3) proj = tm @ W   (WMMA, B staged through LDS double buffer)
  gemm1_proj<<<dim3(kLT / 128, kD / 64, kB), 256, 0, stream>>>(tm_bf, wt_bf, proj_bf);

  // 4) S = proj @ fm^T -> row/col max (WMMA + encoded atomic max)
  gemm2_scoremax<<<dim3(kLT / 128, kLF / 64, kB), 256, 0, stream>>>(proj_bf, fm_bf,
                                                                   rowmax, colmax);
  // 5) tanh + bias + softmax
  biased_softmax<<<dim3(kB, 2), 256, 0, stream>>>(rowmax, colmax, mask_t, mask_f,
                                                  alpha_t, alpha_f);
  // 6) weighted sums -> out
  weighted_sum<<<kB, 512, 0, stream>>>(tm_bf, fm_bf, alpha_t, alpha_f, out);
}
